// CharOffsetAttention_36077725286425
// MI455X (gfx1250) — compile-verified, matrix operand order checked by reading, not run on hardware
//
#include <hip/hip_runtime.h>

// ---------------------------------------------------------------------------
// CharOffsetAttention for MI455X (gfx1250, wave32, WMMA).
// Pipeline: f32->bf16 | GEMM(QKV) | RoPE+relayout | flash-attn | GEMM(out).
// All contractions use v_wmma_f32_16x16x32_bf16 (fp32 accumulate).
// Workspace requirement: ~137 MB.
// ---------------------------------------------------------------------------

#define DEV __device__ __forceinline__

typedef __bf16 bf16;
typedef __attribute__((ext_vector_type(16))) __bf16 v16bf;
typedef __attribute__((ext_vector_type(8)))  __bf16 v8bf;
typedef __attribute__((ext_vector_type(8)))  float  v8f;

// Problem constants (match reference).
constexpr int Bn  = 4;
constexpr int Tn  = 2048;
constexpr int Dn  = 1024;
constexpr int Hn  = 16;
constexpr int HDn = 64;          // head dim
constexpr int Mn  = Bn * Tn;     // 8192 rows

// ---- workspace layout (bytes, 256-aligned sections) ----
constexpr size_t SZ_ACT = (size_t)Mn * Dn * 2;   // 16 MiB (bf16 activation)
constexpr size_t SZ_W   = (size_t)Dn * Dn * 2;   // 2 MiB  (bf16 weight)
constexpr size_t OFF_XB   = 0;
constexpr size_t OFF_WQB  = OFF_XB  + SZ_ACT;
constexpr size_t OFF_WKB  = OFF_WQB + SZ_W;
constexpr size_t OFF_WVB  = OFF_WKB + SZ_W;
constexpr size_t OFF_WOB  = OFF_WVB + SZ_W;
constexpr size_t OFF_QB   = OFF_WOB + SZ_W;      // [M, D] bf16
constexpr size_t OFF_KB   = OFF_QB  + SZ_ACT;
constexpr size_t OFF_VB   = OFF_KB  + SZ_ACT;
constexpr size_t OFF_QR   = OFF_VB  + SZ_ACT;    // [B,H,T,64] bf16 (rotated)
constexpr size_t OFF_KR   = OFF_QR  + SZ_ACT;
constexpr size_t OFF_VR   = OFF_KR  + SZ_ACT;
constexpr size_t OFF_AT   = OFF_VR  + SZ_ACT;    // attention out [M, D] bf16

// ---------------------------------------------------------------------------
DEV v8f wmma_bf16(v16bf a, v16bf b, v8f c) {
  // 8 args: (neg_a, A, neg_b, B, c_mod, C, reuse_a, reuse_b)
  return __builtin_amdgcn_wmma_f32_16x16x32_bf16(false, a, false, b,
                                                 (short)0, c, false, false);
}

// Load a 16x32 A/B fragment from row-major storage with contiguous
// contraction dim.  wave32 layout: row = lane&15; lane half selects K-halves;
// element e -> k = (e<8 ? e : e+8) + 8*(lane>>4).  Two 16-byte loads per lane.
DEV v16bf load_frag16(const bf16* base, int stride, int lane) {
  const int row = lane & 15;
  const int kh  = (lane >> 4) << 3;            // 0 or 8
  const bf16* p = base + row * stride + kh;
  v8bf lo = *(const v8bf*)(p);                 // k = kh .. kh+7
  v8bf hi = *(const v8bf*)(p + 16);            // k = kh+16 .. kh+23
  v16bf r;
#pragma unroll
  for (int i = 0; i < 8; ++i) { r[i] = lo[i]; r[i + 8] = hi[i]; }
  return r;
}

DEV v8f vzero8() {
  v8f z;
#pragma unroll
  for (int i = 0; i < 8; ++i) z[i] = 0.0f;
  return z;
}

// ---------------------------------------------------------------------------
// 1) f32 -> bf16 conversion (grid-stride)
__global__ void cvt_f32_bf16_kernel(const float* __restrict__ in,
                                    bf16* __restrict__ out, int n) {
  for (int i = blockIdx.x * blockDim.x + threadIdx.x; i < n;
       i += gridDim.x * blockDim.x)
    out[i] = (bf16)in[i];
}

// ---------------------------------------------------------------------------
// 2) GEMM: C[M,N] = A[M,K] * W[N,K]^T   (A,W bf16; C bf16 or f32)
//    128x128 block tile, 8 waves (4 along M x 2 along N),
//    each wave: 32x64 = 2x4 fragments, K stepped by 32.
template <bool OUT_BF16>
__global__ __launch_bounds__(256) void gemm_bt_kernel(
    const bf16* __restrict__ A, const bf16* __restrict__ W,
    float* __restrict__ Cf, bf16* __restrict__ Cb, int M, int N, int K) {
  __shared__ __align__(16) bf16 sA[128 * 32];
  __shared__ __align__(16) bf16 sB[128 * 32];

  const int tid  = threadIdx.x;
  const int lane = tid & 31;
  const int wid  = tid >> 5;       // 0..7
  const int wm   = wid & 3;        // wave row  (x32 rows)
  const int wn   = wid >> 2;       // wave col  (x64 cols)
  const int bm   = blockIdx.x * 128;
  const int bn   = blockIdx.y * 128;

  v8f acc[2][4];
#pragma unroll
  for (int i = 0; i < 2; ++i)
#pragma unroll
    for (int j = 0; j < 4; ++j) acc[i][j] = vzero8();

  for (int k0 = 0; k0 < K; k0 += 32) {
    // cooperative tile load: 128x32 bf16 (8 KiB) each = 512 x 16B chunks
#pragma unroll
    for (int i = 0; i < 2; ++i) {
      const int c   = tid + i * 256;     // 0..511
      const int row = c >> 2;
      const int cc  = (c & 3) * 8;
      *(uint4*)&sA[row * 32 + cc] =
          *(const uint4*)&A[(size_t)(bm + row) * K + k0 + cc];
      *(uint4*)&sB[row * 32 + cc] =
          *(const uint4*)&W[(size_t)(bn + row) * K + k0 + cc];
    }
    __syncthreads();

    v16bf af[2], bf_[4];
#pragma unroll
    for (int i = 0; i < 2; ++i)
      af[i] = load_frag16(&sA[(wm * 32 + i * 16) * 32], 32, lane);
#pragma unroll
    for (int j = 0; j < 4; ++j)
      bf_[j] = load_frag16(&sB[(wn * 64 + j * 16) * 32], 32, lane);

#pragma unroll
    for (int i = 0; i < 2; ++i)
#pragma unroll
      for (int j = 0; j < 4; ++j) acc[i][j] = wmma_bf16(af[i], bf_[j], acc[i][j]);
    __syncthreads();
  }

  // C/D layout: VGPR r, lanes 0-15 -> M=r ; lanes 16-31 -> M=r+8 ; N=lane&15
  const int half = lane >> 4;
  const int col  = lane & 15;
#pragma unroll
  for (int i = 0; i < 2; ++i)
#pragma unroll
    for (int j = 0; j < 4; ++j)
#pragma unroll
      for (int r = 0; r < 8; ++r) {
        const int row = bm + wm * 32 + i * 16 + r + 8 * half;
        const int cc  = bn + wn * 64 + j * 16 + col;
        if (OUT_BF16) Cb[(size_t)row * N + cc] = (bf16)acc[i][j][r];
        else          Cf[(size_t)row * N + cc] = acc[i][j][r];
      }
}

// ---------------------------------------------------------------------------
// 3) RoPE (gathered by position_ids) + relayout [B,T,H,hd] -> [B,H,T,hd]
__global__ __launch_bounds__(256) void rope_relayout_kernel(
    const bf16* __restrict__ q, const bf16* __restrict__ k,
    const bf16* __restrict__ v, const int* __restrict__ pos_ids,
    const float* __restrict__ fcos, const float* __restrict__ fsin,
    bf16* __restrict__ qr, bf16* __restrict__ kr, bf16* __restrict__ vr) {
  const int idx = blockIdx.x * blockDim.x + threadIdx.x;  // one per (b,h,t)
  if (idx >= Bn * Hn * Tn) return;
  const int t   = idx % Tn;
  const int h   = (idx / Tn) % Hn;
  const int b   = idx / (Tn * Hn);
  const int pos = pos_ids[b * Tn + t];

  const size_t src = ((size_t)(b * Tn + t)) * Dn + h * HDn;
  const size_t dst = (((size_t)(b * Hn + h)) * Tn + t) * HDn;
  const float* cp = fcos + (size_t)pos * (HDn / 2);
  const float* sp = fsin + (size_t)pos * (HDn / 2);

#pragma unroll 8
  for (int j = 0; j < HDn / 2; ++j) {
    const float c = cp[j], s = sp[j];
    const float qre = (float)q[src + 2 * j], qim = (float)q[src + 2 * j + 1];
    const float kre = (float)k[src + 2 * j], kim = (float)k[src + 2 * j + 1];
    qr[dst + 2 * j]     = (bf16)(qre * c - qim * s);
    qr[dst + 2 * j + 1] = (bf16)(qre * s + qim * c);
    kr[dst + 2 * j]     = (bf16)(kre * c - kim * s);
    kr[dst + 2 * j + 1] = (bf16)(kre * s + kim * c);
    vr[dst + 2 * j]     = v[src + 2 * j];
    vr[dst + 2 * j + 1] = v[src + 2 * j + 1];
  }
}

// ---------------------------------------------------------------------------
// 4) Flash attention (causal). grid = (B*H, T/64). 4 waves x 16 query rows.
//    Streams 32-key blocks: QK^T (4 wmma), online softmax, P*V (4 wmma).
__global__ __launch_bounds__(128) void flash_attn_kernel(
    const bf16* __restrict__ qr, const bf16* __restrict__ kr,
    const bf16* __restrict__ vr, bf16* __restrict__ attnb) {
  __shared__ __align__(16) bf16 sK[32 * 64];        // [key][d]
  __shared__ __align__(16) bf16 sVt[64 * 32];       // [d][key] (transposed)
  __shared__ __align__(16) bf16 sP[4 * 16 * 32];    // per-wave P scratch

  const int tid  = threadIdx.x;
  const int lane = tid & 31;
  const int wid  = tid >> 5;
  const int bh   = blockIdx.x;                 // b*H + h
  const int b    = bh >> 4;
  const int h    = bh & 15;
  const size_t head = (size_t)bh * Tn * HDn;
  const int q0   = blockIdx.y * 64 + wid * 16; // first query row of this wave
  const int half = lane >> 4;
  const int col  = lane & 15;
  const float scale = 0.125f;                  // 1/sqrt(64)

  // Q fragments (d 0..31 and 32..63) straight from global (read once).
  v16bf Qf[2];
  {
    const bf16* qp = qr + head + (size_t)(q0 + col) * HDn + (half << 3);
#pragma unroll
    for (int i = 0; i < 2; ++i) {
      v8bf lo = *(const v8bf*)(qp + i * 32);
      v8bf hi = *(const v8bf*)(qp + i * 32 + 16);
#pragma unroll
      for (int e = 0; e < 8; ++e) { Qf[i][e] = lo[e]; Qf[i][e + 8] = hi[e]; }
    }
  }

  float mrow[8], lrow[8];
  v8f o[4];
#pragma unroll
  for (int r = 0; r < 8; ++r) { mrow[r] = -INFINITY; lrow[r] = 0.0f; }
#pragma unroll
  for (int n = 0; n < 4; ++n) o[n] = vzero8();

  const int kend = blockIdx.y * 64 + 64;       // causal bound (exclusive)
  for (int kb = 0; kb < kend; kb += 32) {
    // --- stage K tile [32,64] and transposed V tile ---
#pragma unroll
    for (int i = 0; i < 2; ++i) {
      const int c   = tid + i * 128;           // 0..255 16B chunks
      const int row = c >> 3;
      const int cc  = (c & 7) * 8;
      *(uint4*)&sK[row * 64 + cc] =
          *(const uint4*)&kr[head + (size_t)(kb + row) * HDn + cc];
      v8bf vv = *(const v8bf*)&vr[head + (size_t)(kb + row) * HDn + cc];
#pragma unroll
      for (int e = 0; e < 8; ++e) sVt[(cc + e) * 32 + row] = vv[e];
    }
    __syncthreads();

    // --- S = Q K^T for two 16-key groups (chained K=32 wmma pairs) ---
    v8f sF[2];
#pragma unroll
    for (int g = 0; g < 2; ++g) {
      v16bf K0 = load_frag16(&sK[(g * 16) * 64 + 0], 64, lane);
      v16bf K1 = load_frag16(&sK[(g * 16) * 64 + 32], 64, lane);
      v8f t = wmma_bf16(Qf[0], K0, vzero8());
      sF[g]  = wmma_bf16(Qf[1], K1, t);
    }

    // --- scale + causal mask + online softmax ---
    float sc[2][8];
#pragma unroll
    for (int g = 0; g < 2; ++g)
#pragma unroll
      for (int r = 0; r < 8; ++r) {
        float v_ = sF[g][r] * scale;
        const int kg = kb + g * 16 + col;
        const int qg = q0 + r + 8 * half;
        sc[g][r] = (kg > qg) ? -INFINITY : v_;
      }

#pragma unroll
    for (int r = 0; r < 8; ++r) {
      float rm = fmaxf(sc[0][r], sc[1][r]);
#pragma unroll
      for (int msk = 1; msk < 16; msk <<= 1)
        rm = fmaxf(rm, __shfl_xor(rm, msk, 32));        // within 16-lane half
      const float mn = fmaxf(mrow[r], rm);
      const float alpha =
          (mrow[r] == -INFINITY) ? 0.0f : __expf(mrow[r] - mn);
      float p0 = (sc[0][r] == -INFINITY) ? 0.0f : __expf(sc[0][r] - mn);
      float p1 = (sc[1][r] == -INFINITY) ? 0.0f : __expf(sc[1][r] - mn);
      float rs = p0 + p1;
#pragma unroll
      for (int msk = 1; msk < 16; msk <<= 1) rs += __shfl_xor(rs, msk, 32);
      lrow[r] = alpha * lrow[r] + rs;
      mrow[r] = mn;
#pragma unroll
      for (int n = 0; n < 4; ++n) o[n][r] *= alpha;
      // P into per-wave LDS scratch (C layout -> row-major [16q][32k])
      bf16* pw = &sP[wid * 512 + (r + 8 * half) * 32];
      pw[col]      = (bf16)p0;
      pw[16 + col] = (bf16)p1;
    }

    // --- O += P V  (A-frag from LDS scratch, B-frags from sVt) ---
    v16bf Pf = load_frag16(&sP[wid * 512], 32, lane);
#pragma unroll
    for (int n = 0; n < 4; ++n) {
      v16bf Vf = load_frag16(&sVt[(n * 16) * 32], 32, lane);
      o[n] = wmma_bf16(Pf, Vf, o[n]);
    }
    __syncthreads();
  }

  // --- normalize and store bf16 into [M, D] for the output projection ---
#pragma unroll
  for (int n = 0; n < 4; ++n)
#pragma unroll
    for (int r = 0; r < 8; ++r) {
      const int qg = q0 + r + 8 * half;
      const int d  = n * 16 + col;
      const float val = o[n][r] / lrow[r];
      attnb[((size_t)(b * Tn + qg)) * Dn + h * HDn + d] = (bf16)val;
    }
}

// ---------------------------------------------------------------------------
extern "C" void kernel_launch(void* const* d_in, const int* in_sizes, int n_in,
                              void* d_out, int out_size, void* d_ws,
                              size_t ws_size, hipStream_t stream) {
  const float* x    = (const float*)d_in[0];
  const int*   pos  = (const int*)d_in[1];
  // d_in[2] = mask (causal tril) -- implied by kernel, unused
  const float* wq   = (const float*)d_in[3];
  const float* wk   = (const float*)d_in[4];
  const float* wv   = (const float*)d_in[5];
  const float* wo   = (const float*)d_in[6];
  const float* fcos = (const float*)d_in[7];
  const float* fsin = (const float*)d_in[8];
  float* out = (float*)d_out;

  char* ws = (char*)d_ws;
  bf16* xb  = (bf16*)(ws + OFF_XB);
  bf16* wqb = (bf16*)(ws + OFF_WQB);
  bf16* wkb = (bf16*)(ws + OFF_WKB);
  bf16* wvb = (bf16*)(ws + OFF_WVB);
  bf16* wob = (bf16*)(ws + OFF_WOB);
  bf16* qb  = (bf16*)(ws + OFF_QB);
  bf16* kb  = (bf16*)(ws + OFF_KB);
  bf16* vb  = (bf16*)(ws + OFF_VB);
  bf16* qr  = (bf16*)(ws + OFF_QR);
  bf16* kr  = (bf16*)(ws + OFF_KR);
  bf16* vr  = (bf16*)(ws + OFF_VR);
  bf16* atb = (bf16*)(ws + OFF_AT);

  // 1) convert activations + weights to bf16
  cvt_f32_bf16_kernel<<<2048, 256, 0, stream>>>(x, xb, Mn * Dn);
  cvt_f32_bf16_kernel<<<1024, 256, 0, stream>>>(wq, wqb, Dn * Dn);
  cvt_f32_bf16_kernel<<<1024, 256, 0, stream>>>(wk, wkb, Dn * Dn);
  cvt_f32_bf16_kernel<<<1024, 256, 0, stream>>>(wv, wvb, Dn * Dn);
  cvt_f32_bf16_kernel<<<1024, 256, 0, stream>>>(wo, wob, Dn * Dn);

  // 2) QKV projections (x @ W^T), bf16 out
  const dim3 ggrid(Mn / 128, Dn / 128);
  gemm_bt_kernel<true><<<ggrid, 256, 0, stream>>>(xb, wqb, nullptr, qb, Mn, Dn, Dn);
  gemm_bt_kernel<true><<<ggrid, 256, 0, stream>>>(xb, wkb, nullptr, kb, Mn, Dn, Dn);
  gemm_bt_kernel<true><<<ggrid, 256, 0, stream>>>(xb, wvb, nullptr, vb, Mn, Dn, Dn);

  // 3) RoPE + relayout to [B,H,T,hd]
  rope_relayout_kernel<<<(Bn * Hn * Tn + 255) / 256, 256, 0, stream>>>(
      qb, kb, vb, pos, fcos, fsin, qr, kr, vr);

  // 4) causal flash attention -> bf16 [M, D]
  flash_attn_kernel<<<dim3(Bn * Hn, Tn / 64), 128, 0, stream>>>(qr, kr, vr, atb);

  // 5) output projection (attn @ wo^T) -> fp32 d_out
  gemm_bt_kernel<false><<<ggrid, 256, 0, stream>>>(atb, wob, out, nullptr, Mn, Dn, Dn);
}
